// STFT_53807350284701
// MI455X (gfx1250) — matrix operand order repaired and checked by reading, not compiled, not run on hardware
//
#include <hip/hip_runtime.h>
#include <math.h>

typedef _Float16 v16h __attribute__((ext_vector_type(16)));
typedef _Float16 v8h  __attribute__((ext_vector_type(8)));
typedef _Float16 v4h  __attribute__((ext_vector_type(4)));
typedef float    v8f  __attribute__((ext_vector_type(8)));
typedef float    v4f  __attribute__((ext_vector_type(4)));

#define FILTER_LEN 1024
#define HOP        512
#define CUTOFF     513
#define NFRAMES    4095          // (2097152 - 1024)/512 + 1
#define T_LEN      2097152
#define LDS_STRIDE 1032          // halves per row: 1024 + 8 pad (2064 B -> bank-conflict-free b128)

static __device__ __forceinline__ v16h cvt16(v4f a, v4f b, v4f c, v4f d) {
    v16h r;
    r[0]  = (_Float16)a.x; r[1]  = (_Float16)a.y; r[2]  = (_Float16)a.z; r[3]  = (_Float16)a.w;
    r[4]  = (_Float16)b.x; r[5]  = (_Float16)b.y; r[6]  = (_Float16)b.z; r[7]  = (_Float16)b.w;
    r[8]  = (_Float16)c.x; r[9]  = (_Float16)c.y; r[10] = (_Float16)c.z; r[11] = (_Float16)c.w;
    r[12] = (_Float16)d.x; r[13] = (_Float16)d.y; r[14] = (_Float16)d.z; r[15] = (_Float16)d.w;
    return r;
}

// Grid: x = frame blocks (128 frames each), y = channel blocks (16 channels), z = batch.
// Block: 256 threads = 8 waves; wave w owns frames [blk*128 + w*16, +16).
// Each wave runs two WMMA chains (real/imag basis vs shared frame fragment).
__global__ __launch_bounds__(256) void stft_mag_wmma(
    const float* __restrict__ x,      // (16, 2097152)
    const float* __restrict__ basis,  // (1026, 1024)
    float* __restrict__ out)          // (16, 513, 4095)
{
    __shared__ _Float16 sB[32 * LDS_STRIDE];   // rows 0-15: real basis, 16-31: imag basis (f16)

    const int tid  = threadIdx.x;
    const int lane = tid & 31;
    const int wave = tid >> 5;
    const int hi   = lane >> 4;     // lane group 0/1
    const int ln   = lane & 15;

    const int c0     = blockIdx.y * 16;                 // channel tile base
    const int b      = blockIdx.z;                      // batch
    const int f_base = blockIdx.x * 128 + wave * 16;    // this wave's frame tile base

    // ---- Stage basis tile (32 rows x 1024 cols) f32 -> f16 into LDS ----
    // 8192 float4 chunks spread over 256 threads (coalesced b128 loads, b64 LDS stores).
    for (int v = tid; v < 32 * 256; v += 256) {
        const int row  = v >> 8;            // 0..31
        const int col4 = (v & 255) << 2;    // 0,4,...,1020
        const int chRel = c0 + ((row < 16) ? row : (row - 16));
        const bool ok   = chRel < CUTOFF;
        const int  ch   = (row < 16) ? chRel : (chRel + CUTOFF);  // imag rows offset by 513
        v4f f;
        if (ok) {
            f = *(const v4f*)(basis + (size_t)ch * FILTER_LEN + col4);
        } else {
            f.x = 0.f; f.y = 0.f; f.z = 0.f; f.w = 0.f;
        }
        v4h h;
        h[0] = (_Float16)f.x; h[1] = (_Float16)f.y;
        h[2] = (_Float16)f.z; h[3] = (_Float16)f.w;
        *(v4h*)(&sB[row * LDS_STRIDE + col4]) = h;
    }
    __syncthreads();

    // ---- Per-lane pointers ----
    // B fragment (frames): lane = column N = frame f_base+ln; lane group selects K half.
    int frame_ld = f_base + ln;
    if (frame_ld > NFRAMES - 1) frame_ld = NFRAMES - 1;   // clamp; masked at store
    const float* xrow = x + (size_t)b * T_LEN + (size_t)frame_ld * HOP + (hi ? 8 : 0);

    // A fragments (basis from LDS): lane row = M = ln; K groups per documented 16x32 layout.
    const _Float16* aRe = &sB[ln * LDS_STRIDE + (hi ? 8 : 0)];
    const _Float16* aIm = aRe + 16 * LDS_STRIDE;

    v8f acc_re = {};
    v8f acc_im = {};

    // ---- K loop: 1024 in steps of 32, unrolled x4 for latency hiding ----
    #pragma unroll 4
    for (int k0 = 0; k0 < FILTER_LEN; k0 += 32) {
        // Shared B fragment: two runs of 8 consecutive f32 samples (k0+8h.., k0+16+8h..) -> f16
        v4f f0 = *(const v4f*)(xrow + k0 + 0);
        v4f f1 = *(const v4f*)(xrow + k0 + 4);
        v4f f2 = *(const v4f*)(xrow + k0 + 16);
        v4f f3 = *(const v4f*)(xrow + k0 + 20);
        v16h bf = cvt16(f0, f1, f2, f3);

        // A fragments from LDS (two 16B ds_load_b128 each, conflict-free)
        v8h rlo = *(const v8h*)(aRe + k0);
        v8h rhi = *(const v8h*)(aRe + k0 + 16);
        v8h ilo = *(const v8h*)(aIm + k0);
        v8h ihi = *(const v8h*)(aIm + k0 + 16);
        v16h aRef = __builtin_shufflevector(rlo, rhi, 0,1,2,3,4,5,6,7,8,9,10,11,12,13,14,15);
        v16h aImf = __builtin_shufflevector(ilo, ihi, 0,1,2,3,4,5,6,7,8,9,10,11,12,13,14,15);

        acc_re = __builtin_amdgcn_wmma_f32_16x16x32_f16(
            false, aRef, false, bf, (short)0, acc_re, false, false);
        acc_im = __builtin_amdgcn_wmma_f32_16x16x32_f16(
            false, aImf, false, bf, (short)0, acc_im, false, false);
    }

    // ---- Epilogue: magnitude + coalesced stores along frame axis ----
    const int frame = f_base + ln;                 // unclamped for masking
    const bool fok  = frame < NFRAMES;
    float* obase = out + (size_t)b * ((size_t)CUTOFF * NFRAMES) + frame;

    #pragma unroll
    for (int i = 0; i < 8; ++i) {
        const int ch = c0 + i + hi * 8;            // D: M = i + 8*laneGroup
        const float re = acc_re[i];
        const float im = acc_im[i];
        // raw v_sqrt_f32 (~1 ulp): avoids the IEEE software-sqrt expansion
        const float m  = __builtin_amdgcn_sqrtf(re * re + im * im);
        if (fok && ch < CUTOFF) {
            obase[(size_t)ch * NFRAMES] = m;       // lanes 0..15 -> 16 consecutive floats
        }
    }
}

extern "C" void kernel_launch(void* const* d_in, const int* in_sizes, int n_in,
                              void* d_out, int out_size, void* d_ws, size_t ws_size,
                              hipStream_t stream) {
    (void)in_sizes; (void)n_in; (void)out_size; (void)d_ws; (void)ws_size;
    const float* x     = (const float*)d_in[0];   // (16, 2097152) f32
    const float* basis = (const float*)d_in[1];   // (1026, 1024)  f32
    float* out         = (float*)d_out;           // (16, 513, 4095) f32

    dim3 block(256, 1, 1);
    dim3 grid((NFRAMES + 127) / 128,              // 32 frame blocks
              (CUTOFF + 15) / 16,                 // 33 channel blocks
              16);                                // batches
    hipLaunchKernelGGL(stft_mag_wmma, grid, block, 0, stream, x, basis, out);
}
